// LSTM_8847632629943
// MI455X (gfx1250) — matrix operand order repaired and checked by reading, not compiled, not run on hardware
//
#include <hip/hip_runtime.h>
#include <hip/hip_bf16.h>

#define T_STEPS 64
#define BATCH   1024
#define LAYERS  10
#define HID     256

typedef __attribute__((ext_vector_type(16))) __bf16 v16bf;
typedef __attribute__((ext_vector_type(8)))  __bf16 v8bf;
typedef __attribute__((ext_vector_type(8)))  float  v8f;

__device__ __forceinline__ unsigned short f32_to_bf16_bits(float f) {
    union { float f; unsigned int u; } x; x.f = f;
    unsigned int r = x.u + 0x7FFFu + ((x.u >> 16) & 1u);   // round-to-nearest-even
    return (unsigned short)(r >> 16);
}

__device__ __forceinline__ float sigmoidf_(float x) {
    return 1.0f / (1.0f + __expf(-x));
}

__device__ __forceinline__ void wait_async0() {
    asm volatile("s_wait_asynccnt 0x0" ::: "memory");
}

template <typename T>
__device__ __forceinline__ unsigned lds_off_of(T* p) {
    // generic pointer to LDS: low 32 bits are the LDS byte offset
    return (unsigned)(unsigned long long)(size_t)p;
}

// ---------------------------------------------------------------------------
// Prep kernels
// ---------------------------------------------------------------------------
__global__ void lstm_zero_u32(unsigned int* __restrict__ p, int n) {
    int i = blockIdx.x * blockDim.x + threadIdx.x;
    if (i < n) p[i] = 0u;
}

// elementwise f32 -> bf16 (inputs)
__global__ void lstm_conv_x(const float* __restrict__ in, unsigned short* __restrict__ out) {
    int i = blockIdx.x * 256 + threadIdx.x;
    out[i] = f32_to_bf16_bits(in[i]);
}

// weights: [l][g][i][h] f32 -> [l][g][h][i] bf16  (transpose + convert)
__global__ void lstm_conv_w(const float* __restrict__ in, unsigned short* __restrict__ out) {
    int o  = blockIdx.x * 256 + threadIdx.x;       // L*4*H*H total, exact multiple
    int i  = o & (HID - 1);
    int h  = (o >> 8) & (HID - 1);
    int gl = o >> 16;                              // l*4 + g
    out[o] = f32_to_bf16_bits(in[(gl * HID + i) * HID + h]);
}

// ---------------------------------------------------------------------------
// LSTM cell: gates = X@WihT + Hprev@WhhT + b ; elementwise -> Hout(bf16), C(f32)
//   grid  = (BATCH/64, HID/32), block = 64 (2 wave32s)
//   WG tile: 64 batch rows x 32 h-cols x 4 gates; wave w -> rows [w*32, w*32+32)
//   Each wave: 2 A-frags x (4 gates x 2 ntiles) = 16 WMMA per k-step.
//   A/B tiles staged via async global->LDS (GVS: saddr + invariant vaddr +
//   compile-time offset immediate), double-buffered.
// ---------------------------------------------------------------------------
#define LDS_STRIDE 40                 // halves per row: 80B, 16B aligned, spreads banks
#define A_ROWS 64
#define B_ROWS 128                    // 4 gates x 32 n
#define A_BUF_B (A_ROWS * LDS_STRIDE * 2)
#define B_BUF_B (B_ROWS * LDS_STRIDE * 2)

__launch_bounds__(64)
__global__ void lstm_cell(const unsigned short* __restrict__ Xbf,    // [B][H] bf16
                          const unsigned short* __restrict__ Hbf,    // [B][H] bf16 (prev t)
                          unsigned short* __restrict__ Hout,         // [B][H] bf16 (this t)
                          float* __restrict__ HoutF32,               // optional f32 copy (may be null)
                          float* __restrict__ C,                     // [B][H] f32, in-place
                          const unsigned short* __restrict__ WtI,    // [4][H][H] bf16, [g][h][i]
                          const unsigned short* __restrict__ WtH,
                          const float* __restrict__ bih,             // [4][H]
                          const float* __restrict__ bhh)
{
    __shared__ __align__(16) unsigned short A_lds[2][A_ROWS * LDS_STRIDE];
    __shared__ __align__(16) unsigned short B_lds[2][B_ROWS * LDS_STRIDE];

    const int tid  = threadIdx.x;         // 0..63
    const int lane = tid & 31;
    const int w    = tid >> 5;            // wave 0..1
    const int half = lane >> 4;
    const int col  = lane & 15;

    const int mb = blockIdx.x;            // batch tile (64 rows)
    const int h0 = blockIdx.y * 32;       // h-col tile base

    const unsigned a_base0 = lds_off_of(&A_lds[0][0]);
    const unsigned b_base0 = lds_off_of(&B_lds[0][0]);

    // ---- per-lane copy descriptors, invariant across k-steps -------------
    unsigned a_vo[4], a_ld[2][4];         // A: 64 rows x 4 chunks = 256 ; 4/thread
    unsigned b_vo[8], b_ld[2][8];         // B: 128 rows x 4 chunks = 512 ; 8/thread
    #pragma unroll
    for (int it = 0; it < 4; ++it) {
        int chunk = tid + it * 64;
        int row = chunk >> 2, c8 = chunk & 3;
        a_vo[it] = (unsigned)(((mb * 64 + row) * HID + c8 * 8) * 2);
        a_ld[0][it] = a_base0 + row * (LDS_STRIDE * 2) + c8 * 16;
        a_ld[1][it] = a_ld[0][it] + A_BUF_B;
    }
    #pragma unroll
    for (int it = 0; it < 8; ++it) {
        int chunk = tid + it * 64;
        int row = chunk >> 2, c8 = chunk & 3;
        int gate = row >> 5, nl = row & 31;
        b_vo[it] = (unsigned)(((gate * HID + h0 + nl) * HID + c8 * 8) * 2);
        b_ld[0][it] = b_base0 + row * (LDS_STRIDE * 2) + c8 * 16;
        b_ld[1][it] = b_ld[0][it] + B_BUF_B;
    }

    // issue all async copies for (compile-time constant) k-step n
    auto issue = [&](const int n) {
        const unsigned long long sA = (unsigned long long)((n < 8) ? Xbf : Hbf);
        const unsigned long long sW = (unsigned long long)((n < 8) ? WtI : WtH);
        const int imm  = (n & 7) * 64;    // k-advance in bytes (32 bf16)
        const int nbuf = n & 1;
        #pragma unroll
        for (int it = 0; it < 4; ++it)
            asm volatile("global_load_async_to_lds_b128 %0, %1, %2 offset:%3"
                         :: "v"(a_ld[nbuf][it]), "v"(a_vo[it]), "s"(sA), "i"(imm)
                         : "memory");
        #pragma unroll
        for (int it = 0; it < 8; ++it)
            asm volatile("global_load_async_to_lds_b128 %0, %1, %2 offset:%3"
                         :: "v"(b_ld[nbuf][it]), "v"(b_vo[it]), "s"(sW), "i"(imm)
                         : "memory");
    };

    v8f acc[4][2][2];                     // [gate][ntile][msub]
    #pragma unroll
    for (int g = 0; g < 4; ++g)
        #pragma unroll
        for (int nt = 0; nt < 2; ++nt)
            #pragma unroll
            for (int ms = 0; ms < 2; ++ms)
                acc[g][nt][ms] = (v8f){0.f,0.f,0.f,0.f,0.f,0.f,0.f,0.f};

    const int arow0 = w * 32 + col;       // ms=0 A-frag row
    const int arow1 = arow0 + 16;         // ms=1 A-frag row

    issue(0);
    #pragma unroll
    for (int ks = 0; ks < 16; ++ks) {
        wait_async0();          // my copies for step ks have landed
        __syncthreads();        // everyone's copies visible; prior compute done
        if (ks + 1 < 16) issue(ks + 1);   // prefetch next step into other buffer

        const int buf = ks & 1;
        const unsigned short* Ab = &A_lds[buf][0];
        const unsigned short* Bb = &B_lds[buf][0];

        // A fragments (16x32 bf16): lane half0 K{0-7,16-23}, half1 K{8-15,24-31}
        v8bf a0lo = *(const v8bf*)(Ab + arow0 * LDS_STRIDE + half * 8);
        v8bf a0hi = *(const v8bf*)(Ab + arow0 * LDS_STRIDE + 16 + half * 8);
        v16bf a0  = __builtin_shufflevector(a0lo, a0hi, 0,1,2,3,4,5,6,7,8,9,10,11,12,13,14,15);
        v8bf a1lo = *(const v8bf*)(Ab + arow1 * LDS_STRIDE + half * 8);
        v8bf a1hi = *(const v8bf*)(Ab + arow1 * LDS_STRIDE + 16 + half * 8);
        v16bf a1  = __builtin_shufflevector(a1lo, a1hi, 0,1,2,3,4,5,6,7,8,9,10,11,12,13,14,15);

        #pragma unroll
        for (int g = 0; g < 4; ++g) {
            #pragma unroll
            for (int nt = 0; nt < 2; ++nt) {
                // B fragment (32x16 bf16): lane half0 K{0-7,16-23}, half1 K{8-15,24-31}
                const int brow = g * 32 + nt * 16 + col;
                v8bf blo = *(const v8bf*)(Bb + brow * LDS_STRIDE + half * 8);
                v8bf bhi = *(const v8bf*)(Bb + brow * LDS_STRIDE + 16 + half * 8);
                v16bf b  = __builtin_shufflevector(blo, bhi, 0,1,2,3,4,5,6,7,8,9,10,11,12,13,14,15);
                acc[g][nt][0] = __builtin_amdgcn_wmma_f32_16x16x32_bf16(
                    false, a0, false, b, (short)0, acc[g][nt][0], false, false);
                acc[g][nt][1] = __builtin_amdgcn_wmma_f32_16x16x32_bf16(
                    false, a1, false, b, (short)0, acc[g][nt][1], false, false);
            }
        }
    }

    // ---- epilogue: lane holds (m = j + 8*half, n = col) of each 16x16 C tile
    #pragma unroll
    for (int nt = 0; nt < 2; ++nt) {
        const int hg = h0 + nt * 16 + col;
        const float bi = bih[0 * HID + hg] + bhh[0 * HID + hg];
        const float bf = bih[1 * HID + hg] + bhh[1 * HID + hg];
        const float bg = bih[2 * HID + hg] + bhh[2 * HID + hg];
        const float bo = bih[3 * HID + hg] + bhh[3 * HID + hg];
        #pragma unroll
        for (int ms = 0; ms < 2; ++ms) {
            const int m_base = mb * 64 + w * 32 + ms * 16 + half * 8;
            #pragma unroll
            for (int j = 0; j < 8; ++j) {
                const int b = m_base + j;
                const float gi = sigmoidf_(acc[0][nt][ms][j] + bi);
                const float gf = sigmoidf_(acc[1][nt][ms][j] + bf);
                const float gg = tanhf   (acc[2][nt][ms][j] + bg);
                const float go = sigmoidf_(acc[3][nt][ms][j] + bo);
                const float cold = C[(size_t)b * HID + hg];
                const float cn = gf * cold + gi * gg;
                const float hn = go * tanhf(cn);
                C[(size_t)b * HID + hg]    = cn;
                Hout[(size_t)b * HID + hg] = f32_to_bf16_bits(hn);
                if (HoutF32) HoutF32[(size_t)b * HID + hg] = hn;
            }
        }
    }
}

// ---------------------------------------------------------------------------
// Host driver
// ---------------------------------------------------------------------------
extern "C" void kernel_launch(void* const* d_in, const int* in_sizes, int n_in,
                              void* d_out, int out_size, void* d_ws, size_t ws_size,
                              hipStream_t stream) {
    const float* inputs = (const float*)d_in[0];   // [T][B][H]
    const float* W_ih   = (const float*)d_in[1];   // [L][4][H][H]
    const float* W_hh   = (const float*)d_in[2];
    const float* b_ih   = (const float*)d_in[3];   // [L][4][H]
    const float* b_hh   = (const float*)d_in[4];

    const size_t WSZ = (size_t)LAYERS * 4 * HID * HID;   // weight elements per tensor
    const size_t LBH = (size_t)BATCH * HID;
    const size_t XSZ = (size_t)T_STEPS * LBH;

    unsigned short* WtI   = (unsigned short*)d_ws;            // [L][4][H][H] bf16
    unsigned short* WtH   = WtI + WSZ;
    unsigned short* Xbf   = WtH + WSZ;                        // [T][B][H] bf16
    unsigned short* h_bf  = Xbf + XSZ;                        // [2][L][B][H] bf16
    float*          c_st  = (float*)(h_bf + 2 * LAYERS * LBH);// [L][B][H] f32
    float*          hf32  = c_st + LAYERS * LBH;              // [B][H] f32 (last layer)

    // zero h_bf (both parities) + c : contiguous region, as u32
    {
        int n = (int)((2 * LAYERS * LBH * 2 + LAYERS * LBH * 4) / 4);
        lstm_zero_u32<<<(n + 255) / 256, 256, 0, stream>>>((unsigned int*)h_bf, n);
    }
    lstm_conv_x<<<(unsigned)(XSZ / 256), 256, 0, stream>>>(inputs, Xbf);
    lstm_conv_w<<<(unsigned)(WSZ / 256), 256, 0, stream>>>(W_ih, WtI);
    lstm_conv_w<<<(unsigned)(WSZ / 256), 256, 0, stream>>>(W_hh, WtH);

    dim3 grid(BATCH / 64, HID / 32);
    dim3 block(64);
    for (int t = 0; t < T_STEPS; ++t) {
        const unsigned short* h_rd = h_bf + (size_t)(t & 1) * LAYERS * LBH;
        unsigned short*       h_wr = h_bf + (size_t)((t + 1) & 1) * LAYERS * LBH;
        for (int l = 0; l < LAYERS; ++l) {
            const unsigned short* X = (l == 0) ? (Xbf + (size_t)t * LBH)
                                               : (h_wr + (size_t)(l - 1) * LBH);
            lstm_cell<<<grid, block, 0, stream>>>(
                X,
                h_rd + (size_t)l * LBH,
                h_wr + (size_t)l * LBH,
                (l == LAYERS - 1) ? hf32 : (float*)nullptr,
                c_st + (size_t)l * LBH,
                WtI + (size_t)l * 4 * HID * HID,
                WtH + (size_t)l * 4 * HID * HID,
                b_ih + (size_t)l * 4 * HID,
                b_hh + (size_t)l * 4 * HID);
        }
    }
    // final h (f32) of last layer, written at t = T-1
    hipMemcpyAsync(d_out, hf32, LBH * sizeof(float), hipMemcpyDeviceToDevice, stream);
}
